// MrT5LayerSelfAttention_83502754169447
// MI455X (gfx1250) — matrix operand order repaired and checked by reading, not compiled, not run on hardware
//
#include <hip/hip_runtime.h>

// ---------------- problem constants ----------------
#define BATCH 2
#define SEQ   2048
#define DMODEL 1024
#define NHEAD 16
#define KVDIM 64
#define MROWS (BATCH * SEQ)      // 4096
#define LN_EPS 1e-6f

typedef __attribute__((ext_vector_type(16))) _Float16 v16h;
typedef __attribute__((ext_vector_type(8)))  float    v8f;
typedef __attribute__((ext_vector_type(4)))  unsigned int u32x4;
typedef __attribute__((ext_vector_type(4)))  int      i32x4;
typedef __attribute__((ext_vector_type(8)))  int      i32x8;

// ---------------- WMMA fragment coordinate maps (ISA 7.12.2, wave32) --------
__device__ __forceinline__ void a_map(int lane, int e, int& m, int& k) {
    const int half = lane >> 4;
    m = lane & 15;
    const int j = e >> 1, bit = e & 1;
    const int base = (j < 4) ? (2 * j) : (16 + 2 * (j - 4));
    k = base + 8 * half + bit;
}
__device__ __forceinline__ void b_map(int lane, int e, int& k, int& n) {
    k = (e < 8) ? (((lane >> 4) << 3) + e) : (16 + ((lane >> 4) << 3) + (e - 8));
    n = lane & 15;
}
__device__ __forceinline__ void cd_map(int lane, int i, int& m, int& n) {
    m = i + ((lane >> 4) << 3);
    n = lane & 15;
}

// ---------------- Tensor Data Mover: 2-D f16 tile -> LDS (ISA ch.8) ---------
// D# group0: count=1 | lds_addr | global_addr[56:0] | type=2
// D# group1: data_size=2B, pad cfg, tensor_dim0/1, tile_dim0/1, dim0 stride
__device__ __forceinline__ void tdm_load_2d_f16(
    unsigned lds_off, const void* gptr,
    unsigned tensor_d0, unsigned tensor_d1,
    unsigned tile_d0, unsigned tile_d1,
    unsigned stride0_elems,
    unsigned pad_enable, unsigned pad_interval_code, unsigned pad_amount_code) {
    const unsigned long long ga = (unsigned long long)(size_t)gptr;
    u32x4 g0;
    g0[0] = 1u;                                        // count=1, user mode
    g0[1] = lds_off;                                   // lds_addr (bytes)
    g0[2] = (unsigned)(ga & 0xFFFFFFFFu);              // global_addr[31:0]
    g0[3] = (unsigned)((ga >> 32) & 0x01FFFFFFu)       // global_addr[56:32]
          | (2u << 30);                                // type = 2 ("image")
    i32x8 g1;
    g1[0] = (int)((1u << 16)                           // data_size = 2 bytes
                | (pad_enable << 20)
                | (pad_interval_code << 22)
                | (pad_amount_code << 25));
    g1[1] = (int)((tensor_d0 & 0xFFFFu) << 16);        // dim0[15:0] @ bits 63:48
    g1[2] = (int)((tensor_d0 >> 16) | ((tensor_d1 & 0xFFFFu) << 16));
    g1[3] = (int)((tensor_d1 >> 16) | (tile_d0 << 16));
    g1[4] = (int)(tile_d1 & 0xFFFFu);                  // tile_dim1 (tile_dim2=0)
    g1[5] = (int)stride0_elems;                        // dim0_stride[31:0]
    g1[6] = 0;                                         // stride0 hi | stride1 lo
    g1[7] = 0;                                         // stride1 hi
    const i32x4 gz = {0, 0, 0, 0};
#if defined(__clang_major__) && (__clang_major__ >= 23)
    const i32x8 gz8 = {0, 0, 0, 0, 0, 0, 0, 0};
    __builtin_amdgcn_tensor_load_to_lds(g0, g1, gz, gz, gz8, 0);
#else
    __builtin_amdgcn_tensor_load_to_lds(g0, g1, gz, gz, 0);
#endif
}

// ---------------- T5 bidirectional relative-position bucket -----------------
__device__ __forceinline__ int t5_bucket(int rel /* key - query */) {
    const int rb = (rel > 0) ? 16 : 0;
    int rp = rel < 0 ? -rel : rel;
    if (rp < 8) return rb + rp;
    int large = 8 + (int)(logf((float)rp * 0.125f) * (8.0f / logf(16.0f)));
    if (large > 15) large = 15;
    return rb + large;
}

// ---------------- kernel 1: RMS norm -> f16 ---------------------------------
__global__ void rmsnorm_f16_kernel(const float* __restrict__ hs,
                                   const float* __restrict__ w,
                                   _Float16* __restrict__ out) {
    const int row = blockIdx.x;
    const int tid = threadIdx.x;         // 256 threads, 4 floats each
    const float4 x = ((const float4*)(hs + (size_t)row * DMODEL))[tid];
    float ss = x.x * x.x + x.y * x.y + x.z * x.z + x.w * x.w;
#pragma unroll
    for (int m = 1; m < 32; m <<= 1) ss += __shfl_xor(ss, m);
    __shared__ float red[8];
    __shared__ float sscale;
    const int wave = tid >> 5, lane = tid & 31;
    if (lane == 0) red[wave] = ss;
    __syncthreads();
    if (tid == 0) {
        float t = 0.f;
#pragma unroll
        for (int i = 0; i < 8; ++i) t += red[i];
        sscale = rsqrtf(t * (1.0f / DMODEL) + LN_EPS);
    }
    __syncthreads();
    const float sc = sscale;
    const float4 wv = ((const float4*)w)[tid];
    _Float16 o[4] = { (_Float16)(x.x * sc * wv.x), (_Float16)(x.y * sc * wv.y),
                      (_Float16)(x.z * sc * wv.z), (_Float16)(x.w * sc * wv.w) };
    *(uint2*)(out + (size_t)row * DMODEL + tid * 4) = *(uint2*)o;
}

// ---------------- WMMA GEMM mainloop (QKV + out-proj) -----------------------
// Block tile 64(M) x 128(N), K-step 32, 8 waves (2x4), 32x32 per wave.
// A tile arrives via TDM with hardware LDS padding: rows of 32 halfs
// (16 DWORDs, pad_interval code 3) + 8 DWORDs pad (code 7) -> stride 48 halfs.
#define BM 64
#define BN 128
#define BK 32
#define LDA 48
#define LDB 136

__device__ __forceinline__ void gemm_block_tile(
    const _Float16* __restrict__ A,     // [MROWS x 1024] f16 row-major
    const float* __restrict__ Wgt,      // [1024 x 1024] f32 row-major
    int m0, int n0,
    _Float16* ldsA, _Float16* ldsB, v8f acc[2][2]) {
    const int tid  = threadIdx.x;
    const int lane = tid & 31;
    const int wave = tid >> 5;
    const int wm = wave >> 2;           // 0..1
    const int wn = wave & 3;            // 0..3
    const unsigned ldsA_off = (unsigned)(size_t)(void*)ldsA;
    for (int k0 = 0; k0 < 1024; k0 += BK) {
        // A tile 64x32 halfs via Tensor Data Mover (one issue per block)
        if (wave == 0) {
            tdm_load_2d_f16(ldsA_off, A + (size_t)m0 * 1024 + k0,
                            /*tensor_d0=*/1024, /*tensor_d1=*/MROWS,
                            /*tile_d0=*/BK, /*tile_d1=*/BM,
                            /*stride0=*/1024,
                            /*pad_en=*/1, /*interval 16 dw=*/3, /*amount 8 dw=*/7);
        }
        // B tile 32x128 f32 -> f16: 256 threads x 16 floats
        {
            const int r = tid >> 3;
            const int c = (tid & 7) * 16;
            const float4* src = (const float4*)(Wgt + (size_t)(k0 + r) * 1024 + n0 + c);
#pragma unroll
            for (int q = 0; q < 4; ++q) {
                const float4 f = src[q];
                _Float16* dst = ldsB + r * LDB + c + q * 4;
                dst[0] = (_Float16)f.x; dst[1] = (_Float16)f.y;
                dst[2] = (_Float16)f.z; dst[3] = (_Float16)f.w;
            }
        }
        if (k0 + BK < 1024)   // prefetch next W K-slab (global_prefetch_b8)
            __builtin_prefetch(Wgt + (size_t)(k0 + BK + (tid >> 3)) * 1024 + n0, 0, 0);
        __builtin_amdgcn_s_wait_tensorcnt(0);
        __syncthreads();
        v16h af[2], bf[2];
#pragma unroll
        for (int mt = 0; mt < 2; ++mt)
#pragma unroll
            for (int e = 0; e < 16; ++e) {
                int m, k; a_map(lane, e, m, k);
                af[mt][e] = ldsA[(wm * 32 + mt * 16 + m) * LDA + k];
            }
#pragma unroll
        for (int nt = 0; nt < 2; ++nt)
#pragma unroll
            for (int e = 0; e < 16; ++e) {
                int k, n; b_map(lane, e, k, n);
                bf[nt][e] = ldsB[k * LDB + wn * 32 + nt * 16 + n];
            }
#pragma unroll
        for (int mt = 0; mt < 2; ++mt)
#pragma unroll
            for (int nt = 0; nt < 2; ++nt)
                acc[mt][nt] = __builtin_amdgcn_wmma_f32_16x16x32_f16(
                    false, af[mt], false, bf[nt], (short)0, acc[mt][nt], false, false);
        __syncthreads();
    }
}

// ---------------- kernel 2: QKV projection ----------------------------------
// mode 0: scatter as [B,H,S,KV] (Q and V) ; mode 1: scatter as [B,H,KV,S] (K^T)
__global__ void qkv_gemm_kernel(const _Float16* __restrict__ X,
                                const float* __restrict__ Wgt,
                                _Float16* __restrict__ Out, int mode) {
    __shared__ __align__(16) _Float16 ldsA[BM * LDA];
    __shared__ __align__(16) _Float16 ldsB[BK * LDB];
    const int m0 = blockIdx.x * BM;
    const int n0 = blockIdx.y * BN;
    const v8f vz = {0, 0, 0, 0, 0, 0, 0, 0};
    v8f acc[2][2] = {{vz, vz}, {vz, vz}};
    gemm_block_tile(X, Wgt, m0, n0, ldsA, ldsB, acc);
    const int lane = threadIdx.x & 31;
    const int wave = threadIdx.x >> 5;
    const int wm = wave >> 2, wn = wave & 3;
#pragma unroll
    for (int mt = 0; mt < 2; ++mt)
#pragma unroll
        for (int nt = 0; nt < 2; ++nt)
#pragma unroll
            for (int i = 0; i < 8; ++i) {
                int m, n; cd_map(lane, i, m, n);
                const int gm = m0 + wm * 32 + mt * 16 + m;
                const int gn = n0 + wn * 32 + nt * 16 + n;
                const int b = gm >> 11, s = gm & (SEQ - 1);
                const int h = gn >> 6,  kv = gn & (KVDIM - 1);
                const _Float16 val = (_Float16)acc[mt][nt][i];
                if (mode == 0)
                    Out[(((size_t)(b * NHEAD + h)) * SEQ + s) * KVDIM + kv] = val;
                else
                    Out[(((size_t)(b * NHEAD + h)) * KVDIM + kv) * SEQ + s] = val;
            }
}

// ---------------- kernel 3: fused flash attention ---------------------------
// grid (SEQ/16, B*H), one wave per block. K/V tiles arrive via TDM.
__global__ void attn_kernel(const _Float16* __restrict__ Q,   // [B,H,S,KV]
                            const _Float16* __restrict__ Kt,  // [B,H,KV,S]
                            const _Float16* __restrict__ V,   // [B,H,S,KV]
                            const float* __restrict__ rel_emb,  // [32,H]
                            const float* __restrict__ amask,    // [B,S]
                            const float* __restrict__ dgate,    // [B,S]
                            _Float16* __restrict__ Ctx) {       // [B,S,H*KV]
    const int qb = blockIdx.x;
    const int bh = blockIdx.y;
    const int b = bh / NHEAD, h = bh % NHEAD;
    const int lane = threadIdx.x;
    const int half = lane >> 4;
    const int lc = lane & 15;
    const int q0 = qb * 16;
    __shared__ __align__(16) _Float16 ldsP[16 * 40];   // P staging, padded
    __shared__ __align__(16) _Float16 kbuf[KVDIM * 32]; // K^T tile 64(kv) x 32(key)
    __shared__ __align__(16) _Float16 vbuf[32 * KVDIM]; // V tile 32(key) x 64(kv)
    const unsigned kbuf_off = (unsigned)(size_t)(void*)kbuf;
    const unsigned vbuf_off = (unsigned)(size_t)(void*)vbuf;

    v16h qa[2];
#pragma unroll
    for (int f = 0; f < 2; ++f)
#pragma unroll
        for (int e = 0; e < 16; ++e) {
            int m, k; a_map(lane, e, m, k);
            qa[f][e] = Q[((size_t)bh * SEQ + q0 + m) * KVDIM + f * 32 + k];
        }

    const v8f vz = {0, 0, 0, 0, 0, 0, 0, 0};
    v8f O[4] = {vz, vz, vz, vz};
    float rmax[8], rsum[8];
#pragma unroll
    for (int i = 0; i < 8; ++i) { rmax[i] = -1e30f; rsum[i] = 0.f; }

    for (int n0 = 0; n0 < SEQ; n0 += 32) {
        // DMA this key tile: K^T rows kv=0..63, cols n0..n0+31 ; V rows n0..n0+31
        tdm_load_2d_f16(kbuf_off, Kt + (size_t)bh * KVDIM * SEQ + n0,
                        SEQ, KVDIM, 32, KVDIM, SEQ, 0, 0, 0);
        tdm_load_2d_f16(vbuf_off, V + ((size_t)bh * SEQ + n0) * KVDIM,
                        KVDIM, SEQ, KVDIM, 32, KVDIM, 0, 0, 0);
        __builtin_amdgcn_s_wait_tensorcnt(0);

        // ---- scores: 16x32 tile = two 16x16 C tiles, KV reduced in 2 chunks
        v8f c[2] = {vz, vz};
#pragma unroll
        for (int sub = 0; sub < 2; ++sub)
#pragma unroll
            for (int f = 0; f < 2; ++f) {
                v16h kb;
#pragma unroll
                for (int e = 0; e < 16; ++e) {
                    int k, n; b_map(lane, e, k, n);
                    kb[e] = kbuf[(f * 32 + k) * 32 + sub * 16 + n];
                }
                c[sub] = __builtin_amdgcn_wmma_f32_16x16x32_f16(
                    false, qa[f], false, kb, (short)0, c[sub], false, false);
            }
        // ---- bias + mask + delete-gate, online softmax update per row -----
#pragma unroll
        for (int i = 0; i < 8; ++i) {
            const int q = q0 + i + 8 * half;
            float s0 = c[0][i], s1 = c[1][i];
            {
                const int key = n0 + lc;
                s0 += rel_emb[t5_bucket(key - q) * NHEAD + h]
                    + amask[b * SEQ + key] + dgate[b * SEQ + key];
            }
            {
                const int key = n0 + 16 + lc;
                s1 += rel_emb[t5_bucket(key - q) * NHEAD + h]
                    + amask[b * SEQ + key] + dgate[b * SEQ + key];
            }
            float t = fmaxf(s0, s1);
#pragma unroll
            for (int mm = 1; mm < 16; mm <<= 1) t = fmaxf(t, __shfl_xor(t, mm));
            const float nm = fmaxf(rmax[i], t);
            const float sc = __expf(rmax[i] - nm);
            const float p0 = __expf(s0 - nm);
            const float p1 = __expf(s1 - nm);
            float ps = p0 + p1;
#pragma unroll
            for (int mm = 1; mm < 16; mm <<= 1) ps += __shfl_xor(ps, mm);
            rsum[i] = rsum[i] * sc + ps;
            rmax[i] = nm;
#pragma unroll
            for (int t4 = 0; t4 < 4; ++t4) O[t4][i] *= sc;
            const int m = i + 8 * half;
            ldsP[m * 40 + lc]      = (_Float16)p0;
            ldsP[m * 40 + 16 + lc] = (_Float16)p1;
        }
        __syncthreads();
        // ---- P (A-layout from LDS) x V tile: accumulate into O -------------
        v16h pa;
#pragma unroll
        for (int e = 0; e < 16; ++e) {
            int m, k; a_map(lane, e, m, k);
            pa[e] = ldsP[m * 40 + k];
        }
#pragma unroll
        for (int t4 = 0; t4 < 4; ++t4) {
            v16h vb;
#pragma unroll
            for (int e = 0; e < 16; ++e) {
                int k, n; b_map(lane, e, k, n);
                vb[e] = vbuf[k * KVDIM + t4 * 16 + n];
            }
            O[t4] = __builtin_amdgcn_wmma_f32_16x16x32_f16(
                false, pa, false, vb, (short)0, O[t4], false, false);
        }
        __syncthreads();
    }
    // ---- normalize and write context as f16 [B,S,H*KV] ---------------------
#pragma unroll
    for (int i = 0; i < 8; ++i) {
        const float inv = 1.0f / rsum[i];
        const int m = i + 8 * half;
        const size_t base = ((size_t)b * SEQ + q0 + m) * (NHEAD * KVDIM) + h * KVDIM;
#pragma unroll
        for (int t4 = 0; t4 < 4; ++t4)
            Ctx[base + t4 * 16 + lc] = (_Float16)(O[t4][i] * inv);
    }
}

// ---------------- kernel 4: output projection + residual --------------------
__global__ void oproj_gemm_kernel(const _Float16* __restrict__ CtxA,
                                  const float* __restrict__ Wo,
                                  const float* __restrict__ hidden,
                                  float* __restrict__ out) {
    __shared__ __align__(16) _Float16 ldsA[BM * LDA];
    __shared__ __align__(16) _Float16 ldsB[BK * LDB];
    const int m0 = blockIdx.x * BM;
    const int n0 = blockIdx.y * BN;
    const v8f vz = {0, 0, 0, 0, 0, 0, 0, 0};
    v8f acc[2][2] = {{vz, vz}, {vz, vz}};
    gemm_block_tile(CtxA, Wo, m0, n0, ldsA, ldsB, acc);
    const int lane = threadIdx.x & 31;
    const int wave = threadIdx.x >> 5;
    const int wm = wave >> 2, wn = wave & 3;
#pragma unroll
    for (int mt = 0; mt < 2; ++mt)
#pragma unroll
        for (int nt = 0; nt < 2; ++nt)
#pragma unroll
            for (int i = 0; i < 8; ++i) {
                int m, n; cd_map(lane, i, m, n);
                const size_t gm = m0 + wm * 32 + mt * 16 + m;
                const size_t gn = n0 + wn * 32 + nt * 16 + n;
                out[gm * DMODEL + gn] = hidden[gm * DMODEL + gn] + acc[mt][nt][i];
            }
}

// ---------------- host-side launcher ----------------------------------------
extern "C" void kernel_launch(void* const* d_in, const int* in_sizes, int n_in,
                              void* d_out, int out_size, void* d_ws, size_t ws_size,
                              hipStream_t stream) {
    const float* hs    = (const float*)d_in[0];
    const float* amask = (const float*)d_in[1];
    const float* dgate = (const float*)d_in[2];
    const float* wq    = (const float*)d_in[3];
    const float* wk    = (const float*)d_in[4];
    const float* wv    = (const float*)d_in[5];
    const float* wo    = (const float*)d_in[6];
    const float* rel   = (const float*)d_in[7];
    const float* lnw   = (const float*)d_in[8];
    float* out = (float*)d_out;

    char* ws = (char*)d_ws;
    const size_t SEG = (size_t)MROWS * DMODEL * sizeof(_Float16); // 8 MB each
    _Float16* Xf = (_Float16*)(ws);
    _Float16* Qd = (_Float16*)(ws + SEG);
    _Float16* Kd = (_Float16*)(ws + 2 * SEG);
    _Float16* Vd = (_Float16*)(ws + 3 * SEG);
    _Float16* Cd = (_Float16*)(ws + 4 * SEG);

    rmsnorm_f16_kernel<<<dim3(MROWS), dim3(256), 0, stream>>>(hs, lnw, Xf);

    dim3 gg(MROWS / BM, (NHEAD * KVDIM) / BN);
    qkv_gemm_kernel<<<gg, dim3(256), 0, stream>>>(Xf, wq, Qd, 0);
    qkv_gemm_kernel<<<gg, dim3(256), 0, stream>>>(Xf, wk, Kd, 1);
    qkv_gemm_kernel<<<gg, dim3(256), 0, stream>>>(Xf, wv, Vd, 0);

    attn_kernel<<<dim3(SEQ / 16, BATCH * NHEAD), dim3(32), 0, stream>>>(
        Qd, Kd, Vd, rel, amask, dgate, Cd);

    oproj_gemm_kernel<<<dim3(MROWS / BM, DMODEL / BN), dim3(256), 0, stream>>>(
        Cd, wo, hs, out);
}
